// Decoder_77318001262794
// MI455X (gfx1250) — matrix operand order repaired and checked by reading, not compiled, not run on hardware
//
#include <hip/hip_runtime.h>

#define DEV_INLINE __device__ __forceinline__

typedef __attribute__((ext_vector_type(16))) __bf16 bf16x16;
typedef __attribute__((ext_vector_type(8)))  float  f32x8;

struct Frag32B { uint4 lo; uint4 hi; };

DEV_INLINE unsigned short f32_to_bf16(float f) {
    unsigned int u = __float_as_uint(f);
    unsigned int r = u + 0x7FFFu + ((u >> 16) & 1u);
    return (unsigned short)(r >> 16);
}
DEV_INLINE float bf16_to_f32(unsigned short u) {
    return __uint_as_float(((unsigned int)u) << 16);
}
DEV_INLINE float silu_f(float v) { return v / (1.f + __expf(-v)); }
DEV_INLINE float gelu_tanh_f(float v) {
    float t = 0.7978845608028654f * (v + 0.044715f * v * v * v);
    return 0.5f * v * (1.f + tanhf(t));
}

// ---- WMMA fragment loaders (CDNA5 wave32 layouts, v_wmma_f32_16x16x32_bf16) ----
// A (16x32, row-major source): lane holds row (lane&15); halves 0..7 = K kb..kb+7,
// halves 8..15 = K kb+16..kb+23, kb = (lane>>4)*8.  p = row_ptr + kb (+kk).
DEV_INLINE bf16x16 frag_ld_a(const unsigned short* p) {
    const uint4* p4 = reinterpret_cast<const uint4*>(p);
    Frag32B f; f.lo = p4[0]; f.hi = p4[2];
    return __builtin_bit_cast(bf16x16, f);
}
// B (32x16 from W^T stored N x K): lane holds column (lane&15); halves 0..7 = K kb..kb+7,
// halves 8..15 = K kb+16..kb+23, kb = (lane>>4)*8.  p = col_ptr + kb (+kk).
DEV_INLINE bf16x16 frag_ld_b(const unsigned short* p) {
    const uint4* p4 = reinterpret_cast<const uint4*>(p);
    Frag32B f; f.lo = p4[0]; f.hi = p4[2];
    return __builtin_bit_cast(bf16x16, f);
}

// ---- Main bf16 WMMA GEMM ----
// Block tile 128(M) x 128(N), 8 waves as 4(M) x 2(N); wave tile 32x64.
// Register double-buffered K pipeline: loads for step k+32 are in flight
// while the 8 WMMAs of step k execute.
// MODE 0: out = acc + bias                         (f32 store)
// MODE 1: out = silu(acc + bias)                   (f32 store)
// MODE 2: out = gelu_tanh(acc + bias)              (bf16 store)
// MODE 3: out = resid + gate[b][col]*(acc + bias)  (f32 store, b = row>>11)
template <int MODE>
__global__ __launch_bounds__(256) void gemm_bf16_kernel(
    const unsigned short* __restrict__ A,   // M x K bf16, row-major
    const unsigned short* __restrict__ Bt,  // N x K bf16 (W transposed)
    const float* __restrict__ bias,         // N
    float* __restrict__ outF,
    unsigned short* __restrict__ outBF,
    const float* __restrict__ resid,
    const float* __restrict__ gate,
    int M, int N, int K, int gateStride, int gateOfs)
{
    const int lane = threadIdx.x & 31;
    const int wave = threadIdx.x >> 5;
    const int mw = wave & 3;            // 0..3
    const int nw = wave >> 2;           // 0..1
    const int m0 = blockIdx.x * 128 + mw * 32;
    const int n0 = blockIdx.y * 128 + nw * 64;

    const unsigned short* pa[2];
    const unsigned short* pb[4];
    #pragma unroll
    for (int i = 0; i < 2; ++i)
        pa[i] = A + (size_t)(m0 + i * 16 + (lane & 15)) * K + ((lane >> 4) << 3);
    #pragma unroll
    for (int j = 0; j < 4; ++j)
        pb[j] = Bt + (size_t)(n0 + j * 16 + (lane & 15)) * K + ((lane >> 4) << 3);

    f32x8 acc[2][4];
    #pragma unroll
    for (int i = 0; i < 2; ++i)
        #pragma unroll
        for (int j = 0; j < 4; ++j)
            #pragma unroll
            for (int r = 0; r < 8; ++r) acc[i][j][r] = 0.f;

    bf16x16 a[2], b[4];
    #pragma unroll
    for (int i = 0; i < 2; ++i) a[i] = frag_ld_a(pa[i]);
    #pragma unroll
    for (int j = 0; j < 4; ++j) b[j] = frag_ld_b(pb[j]);

    #pragma unroll 2
    for (int kk = 32; kk < K; kk += 32) {
        bf16x16 an[2], bn[4];
        #pragma unroll
        for (int i = 0; i < 2; ++i) an[i] = frag_ld_a(pa[i] + kk);
        #pragma unroll
        for (int j = 0; j < 4; ++j) bn[j] = frag_ld_b(pb[j] + kk);
        __builtin_prefetch(pa[0] + kk + 96, 0, 3);
        __builtin_prefetch(pb[0] + kk + 96, 0, 3);
        #pragma unroll
        for (int i = 0; i < 2; ++i)
            #pragma unroll
            for (int j = 0; j < 4; ++j)
                acc[i][j] = __builtin_amdgcn_wmma_f32_16x16x32_bf16(
                    false, a[i], false, b[j], (short)0, acc[i][j], false, false);
        #pragma unroll
        for (int i = 0; i < 2; ++i) a[i] = an[i];
        #pragma unroll
        for (int j = 0; j < 4; ++j) b[j] = bn[j];
    }
    #pragma unroll
    for (int i = 0; i < 2; ++i)
        #pragma unroll
        for (int j = 0; j < 4; ++j)
            acc[i][j] = __builtin_amdgcn_wmma_f32_16x16x32_bf16(
                false, a[i], false, b[j], (short)0, acc[i][j], false, false);

    // ---- epilogue ----
    const int col_lane = lane & 15;
    const int row_off  = (lane >> 4) << 3;
    #pragma unroll
    for (int i = 0; i < 2; ++i) {
        const int row_base = m0 + i * 16 + row_off;
        #pragma unroll
        for (int j = 0; j < 4; ++j) {
            const int col = n0 + j * 16 + col_lane;
            const float bv = bias[col];
            #pragma unroll
            for (int r = 0; r < 8; ++r) {
                const int row = row_base + r;
                const size_t idx = (size_t)row * N + col;
                float v = acc[i][j][r] + bv;
                if (MODE == 0) {
                    outF[idx] = v;
                } else if (MODE == 1) {
                    outF[idx] = silu_f(v);
                } else if (MODE == 2) {
                    outBF[idx] = f32_to_bf16(gelu_tanh_f(v));
                } else {
                    const int bb = row >> 11; // 2048 rows per batch
                    const float gt = gate[(size_t)bb * gateStride + gateOfs + col];
                    outF[idx] = resid[idx] + gt * v;
                }
            }
        }
    }
}

// ---- Row LayerNorm; optional gamma/beta, optional adaLN modulation ----
template <bool OUT_BF16, bool AFFINE, bool MOD>
__global__ void ln_kernel(const float* __restrict__ X, int N,
                          const float* __restrict__ g, const float* __restrict__ be,
                          const float* __restrict__ modBase, int modStride,
                          int shOfs, int scOfs, int rowsPerB,
                          float* __restrict__ outF, unsigned short* __restrict__ outB)
{
    __shared__ float red0[32];
    __shared__ float red1[32];
    const int row = blockIdx.x;
    const float* xr = X + (size_t)row * N;
    float s = 0.f, s2 = 0.f;
    for (int i = threadIdx.x; i < N; i += blockDim.x) {
        float v = xr[i]; s += v; s2 += v * v;
    }
    #pragma unroll
    for (int o = 16; o > 0; o >>= 1) {
        s  += __shfl_down(s,  o, 32);
        s2 += __shfl_down(s2, o, 32);
    }
    const int wid = threadIdx.x >> 5;
    const int nw  = blockDim.x >> 5;
    if ((threadIdx.x & 31) == 0) { red0[wid] = s; red1[wid] = s2; }
    __syncthreads();
    if (threadIdx.x == 0) {
        float aa = 0.f, b2 = 0.f;
        for (int w = 0; w < nw; ++w) { aa += red0[w]; b2 += red1[w]; }
        red0[0] = aa; red1[0] = b2;
    }
    __syncthreads();
    const float mean = red0[0] / (float)N;
    const float var  = red1[0] / (float)N - mean * mean;
    const float inv  = rsqrtf(var + 1e-5f);
    const int bb = MOD ? (row / rowsPerB) : 0;
    for (int i = threadIdx.x; i < N; i += blockDim.x) {
        float v = (xr[i] - mean) * inv;
        if (AFFINE) v = v * g[i] + be[i];
        if (MOD) {
            const float sc = modBase[(size_t)bb * modStride + scOfs + i];
            const float sh = modBase[(size_t)bb * modStride + shOfs + i];
            v = v * (1.f + sc) + sh;
        }
        if (OUT_BF16) outB[(size_t)row * N + i] = f32_to_bf16(v);
        else          outF[(size_t)row * N + i] = v;
    }
}

// ---- weight convert + transpose: f32 (K x N) -> bf16 (N x K) ----
__global__ void wconv_t_kernel(const float* __restrict__ W, unsigned short* __restrict__ out,
                               int K, int N)
{
    int idx = blockIdx.x * blockDim.x + threadIdx.x;
    if (idx >= K * N) return;
    int n = idx / K, k = idx % K;
    out[(size_t)n * K + k] = f32_to_bf16(W[(size_t)k * N + n]);
}

// ---- x0 = gather(residue by n/8) + atom_latent -> bf16 ----
__global__ void x0_kernel(const float* __restrict__ residue, const float* __restrict__ atom,
                          unsigned short* __restrict__ out)
{
    int idx = blockIdx.x * blockDim.x + threadIdx.x;
    if (idx >= 2 * 2048 * 128) return;
    int d = idx & 127;
    int n = (idx >> 7) & 2047;
    int b = idx >> 18;
    float v = residue[((size_t)(b * 256 + (n >> 3))) * 128 + d] + atom[idx];
    out[idx] = f32_to_bf16(v);
}

// ---- small dense: out[r][j] = sum_k act(in[r][k]) * W[k][j] + bias[j] ----
template <bool SILU>
__global__ void small_mm_kernel(const float* __restrict__ in, int inStride,
                                const float* __restrict__ W, const float* __restrict__ bias,
                                float* __restrict__ out, int R, int K, int N)
{
    int idx = blockIdx.x * blockDim.x + threadIdx.x;
    if (idx >= R * N) return;
    int r = idx / N, j = idx % N;
    float s = bias[j];
    for (int k = 0; k < K; ++k) {
        float v = in[(size_t)r * inStride + k];
        if (SILU) v = silu_f(v);
        s += v * W[(size_t)k * N + j];
    }
    out[(size_t)r * N + j] = s;
}

// ---- RoPE in-place on q and k parts of qkv (B,NA,3,H,48) f32 ----
__global__ void rope_kernel(float* __restrict__ qkv, const float* __restrict__ pos)
{
    int idx = blockIdx.x * blockDim.x + threadIdx.x;
    const int total = 2 * 2048 * 16 * 24;
    if (idx >= total) return;
    int t = idx % 24;
    int h = (idx / 24) & 15;
    int n = (idx / (24 * 16)) & 2047;
    int b = idx / (24 * 16 * 2048);
    int axis = t >> 3, j = t & 7;
    float inv = __powf(10000.f, -(float)j * 0.125f);
    float ang = pos[((size_t)(b * 2048 + n)) * 3 + axis] * inv;
    float c = cosf(ang), s = sinf(ang);
    size_t rq = ((size_t)(b * 2048 + n)) * 2304 + (size_t)h * 48 + 2 * t;
    float x1 = qkv[rq], x2 = qkv[rq + 1];
    qkv[rq]     = x1 * c - x2 * s;
    qkv[rq + 1] = x1 * s + x2 * c;
    size_t rk = rq + 768;
    x1 = qkv[rk]; x2 = qkv[rk + 1];
    qkv[rk]     = x1 * c - x2 * s;
    qkv[rk + 1] = x1 * s + x2 * c;
}

// ---- local-window attention, flash-style, one wave per (b,h,qblock) ----
__global__ __launch_bounds__(32) void attn_kernel(const float* __restrict__ qkv,
                                                  unsigned short* __restrict__ outB)
{
    const int blk = blockIdx.x;          // b*1024 + h*64 + qb
    const int qb = blk & 63;
    const int h  = (blk >> 6) & 15;
    const int b  = blk >> 10;
    const int lane = threadIdx.x;        // query within block
    const int n = qb * 32 + lane;

    float q[48], o[48];
    const size_t qrow = ((size_t)(b * 2048 + n)) * 2304 + (size_t)h * 48;
    #pragma unroll
    for (int d = 0; d < 48; ++d) { q[d] = qkv[qrow + d]; o[d] = 0.f; }

    int j1 = 32 * qb - 48; if (j1 < 0) j1 = 0;
    int j2 = 32 * qb + 80; if (j2 > 2048) j2 = 2048;
    const float scale = rsqrtf(48.f);

    float mM = -3.0e38f, l = 0.f;
    for (int j = j1; j < j2; ++j) {
        const size_t kro = ((size_t)(b * 2048 + j)) * 2304 + 768 + (size_t)h * 48;
        float s = 0.f;
        #pragma unroll
        for (int d = 0; d < 48; ++d) s += q[d] * qkv[kro + d];
        s *= scale;
        const float nm = fmaxf(mM, s);
        const float corr = __expf(mM - nm);
        const float p = __expf(s - nm);
        l = l * corr + p;
        const size_t vro = kro + 768;
        #pragma unroll
        for (int d = 0; d < 48; ++d) o[d] = o[d] * corr + p * qkv[vro + d];
        mM = nm;
    }
    const float invl = 1.f / l;
    const size_t orow = ((size_t)(b * 2048 + n)) * 768 + (size_t)h * 48;
    #pragma unroll
    for (int d = 0; d < 48; ++d) outB[orow + d] = f32_to_bf16(o[d] * invl);
}

// ---- final projection: (4096 x 768 bf16) @ (768 x 3) + b -> f32 ----
__global__ void final_out_kernel(const unsigned short* __restrict__ hF,
                                 const float* __restrict__ W, const float* __restrict__ bias,
                                 float* __restrict__ out)
{
    int idx = blockIdx.x * blockDim.x + threadIdx.x;
    if (idx >= 4096 * 3) return;
    int row = idx / 3, j = idx % 3;
    float s = bias[j];
    const unsigned short* hp = hF + (size_t)row * 768;
    for (int k = 0; k < 768; ++k) s += bf16_to_f32(hp[k]) * W[k * 3 + j];
    out[idx] = s;
}

extern "C" void kernel_launch(void* const* d_in, const int* in_sizes, int n_in,
                              void* d_out, int out_size, void* d_ws, size_t ws_size,
                              hipStream_t stream)
{
    (void)in_sizes; (void)n_in; (void)out_size; (void)ws_size;

    const float* residue  = (const float*)d_in[0];
    const float* atom     = (const float*)d_in[1];
    const float* adaLN    = (const float*)d_in[2];
    const float* rope_pos = (const float*)d_in[3];
    // d_in[4] atom_to_token == one_hot(n/8): implemented as gather in x0_kernel
    const float* p1_W  = (const float*)d_in[5];
    const float* p1_b  = (const float*)d_in[6];
    const float* p1_g  = (const float*)d_in[7];
    const float* p1_be = (const float*)d_in[8];
    const float* p2_W  = (const float*)d_in[9];
    const float* p2_b  = (const float*)d_in[10];
    const float* ap_W  = (const float*)d_in[11];
    const float* ap_b  = (const float*)d_in[12];
    const float* ap_g  = (const float*)d_in[13];
    const float* ap_be = (const float*)d_in[14];
    const float* ada_W = (const float*)d_in[15];
    const float* ada_b = (const float*)d_in[16];
    const float* qkv_W = (const float*)d_in[17];
    const float* qkv_b = (const float*)d_in[18];
    const float* o_W   = (const float*)d_in[19];
    const float* o_b   = (const float*)d_in[20];
    const float* m1_W  = (const float*)d_in[21];
    const float* m1_b  = (const float*)d_in[22];
    const float* m2_W  = (const float*)d_in[23];
    const float* m2_b  = (const float*)d_in[24];
    const float* fl_ada_W = (const float*)d_in[25];
    const float* fl_ada_b = (const float*)d_in[26];
    const float* fl_out_W = (const float*)d_in[27];
    const float* fl_out_b = (const float*)d_in[28];

    const int Bz = 2, NA = 2048, DA = 128, D = 768;
    const int M = Bz * NA; // 4096

    // --- workspace bump allocator (256B aligned) ---
    char* base = (char*)d_ws;
    size_t off = 0;
    auto alloc = [&](size_t bytes) -> void* {
        void* p = base + off;
        off += (bytes + 255) & ~(size_t)255;
        return p;
    };
    unsigned short* wt_p1 = (unsigned short*)alloc((size_t)128 * 128 * 2);
    unsigned short* wt_p2 = (unsigned short*)alloc((size_t)768 * 128 * 2);
    unsigned short* wt_qkv[2], *wt_o[2], *wt_m1[2], *wt_m2[2];
    for (int l = 0; l < 2; ++l) {
        wt_qkv[l] = (unsigned short*)alloc((size_t)2304 * 768 * 2);
        wt_o[l]   = (unsigned short*)alloc((size_t)768 * 768 * 2);
        wt_m1[l]  = (unsigned short*)alloc((size_t)3072 * 768 * 2);
        wt_m2[l]  = (unsigned short*)alloc((size_t)768 * 3072 * 2);
    }
    unsigned short* x0    = (unsigned short*)alloc((size_t)M * DA * 2);
    float*          h1    = (float*)alloc((size_t)M * DA * 4);
    unsigned short* h1ln  = (unsigned short*)alloc((size_t)M * DA * 2);
    float*          x     = (float*)alloc((size_t)M * D * 4);
    float*          ctmp  = (float*)alloc((size_t)Bz * D * 4);
    float*          cbuf  = (float*)alloc((size_t)Bz * D * 4);
    float*          mods  = (float*)alloc((size_t)Bz * 4608 * 4);
    unsigned short* hmod  = (unsigned short*)alloc((size_t)M * D * 2);
    float*          qkv   = (float*)alloc((size_t)M * 2304 * 4);
    unsigned short* attnO = (unsigned short*)alloc((size_t)M * D * 2);
    unsigned short* mlp   = (unsigned short*)alloc((size_t)M * 3072 * 2);
    float*          fm    = (float*)alloc((size_t)Bz * 1536 * 4);
    unsigned short* hF    = (unsigned short*)alloc((size_t)M * D * 2);

    auto wconv = [&](const float* W, unsigned short* dst, int K, int N) {
        int tot = K * N;
        wconv_t_kernel<<<(tot + 255) / 256, 256, 0, stream>>>(W, dst, K, N);
    };

    // --- weight prep (deterministic, redone every call) ---
    wconv(p1_W, wt_p1, 128, 128);
    wconv(p2_W, wt_p2, 128, 768);
    for (int l = 0; l < 2; ++l) {
        wconv(qkv_W + (size_t)l * 768 * 2304, wt_qkv[l], 768, 2304);
        wconv(o_W   + (size_t)l * 768 * 768,  wt_o[l],   768, 768);
        wconv(m1_W  + (size_t)l * 768 * 3072, wt_m1[l],  768, 3072);
        wconv(m2_W  + (size_t)l * 3072 * 768, wt_m2[l],  3072, 768);
    }

    // --- stem: x = LN(silu(x0@p1+b)) @ p2 + b ---
    x0_kernel<<<(M * DA + 255) / 256, 256, 0, stream>>>(residue, atom, x0);
    gemm_bf16_kernel<1><<<dim3(M / 128, DA / 128), 256, 0, stream>>>(
        x0, wt_p1, p1_b, h1, nullptr, nullptr, nullptr, M, DA, DA, 0, 0);
    ln_kernel<true, true, false><<<M, 128, 0, stream>>>(
        h1, DA, p1_g, p1_be, nullptr, 0, 0, 0, NA, nullptr, h1ln);
    gemm_bf16_kernel<0><<<dim3(M / 128, D / 128), 256, 0, stream>>>(
        h1ln, wt_p2, p2_b, x, nullptr, nullptr, nullptr, M, D, DA, 0, 0);

    // --- c = LN(adaLN @ ap_W + ap_b, g, be) ---
    small_mm_kernel<false><<<(Bz * D + 255) / 256, 256, 0, stream>>>(
        adaLN, DA, ap_W, ap_b, ctmp, Bz, DA, D);
    ln_kernel<false, true, false><<<Bz, 256, 0, stream>>>(
        ctmp, D, ap_g, ap_be, nullptr, 0, 0, 0, 1, cbuf, nullptr);

    // --- transformer layers ---
    for (int l = 0; l < 2; ++l) {
        small_mm_kernel<true><<<(Bz * 4608 + 255) / 256, 256, 0, stream>>>(
            cbuf, D, ada_W + (size_t)l * D * 4608, ada_b + (size_t)l * 4608,
            mods, Bz, D, 4608);
        // hmod = LN(x)*(1+sc1)+sh1
        ln_kernel<true, false, true><<<M, 256, 0, stream>>>(
            x, D, nullptr, nullptr, mods, 4608, 0, 768, NA, nullptr, hmod);
        // qkv
        gemm_bf16_kernel<0><<<dim3(M / 128, 2304 / 128), 256, 0, stream>>>(
            hmod, wt_qkv[l], qkv_b + (size_t)l * 2304, qkv, nullptr, nullptr, nullptr,
            M, 2304, D, 0, 0);
        rope_kernel<<<(2 * 2048 * 16 * 24 + 255) / 256, 256, 0, stream>>>(qkv, rope_pos);
        attn_kernel<<<Bz * 16 * 64, 32, 0, stream>>>(qkv, attnO);
        // x = x + g1 * (attnO @ o_W + o_b)
        gemm_bf16_kernel<3><<<dim3(M / 128, D / 128), 256, 0, stream>>>(
            attnO, wt_o[l], o_b + (size_t)l * D, x, nullptr, x, mods,
            M, D, D, 4608, 1536);
        // hmod = LN(x)*(1+sc2)+sh2
        ln_kernel<true, false, true><<<M, 256, 0, stream>>>(
            x, D, nullptr, nullptr, mods, 4608, 2304, 3072, NA, nullptr, hmod);
        // mlp = gelu(hmod @ m1 + b)
        gemm_bf16_kernel<2><<<dim3(M / 128, 3072 / 128), 256, 0, stream>>>(
            hmod, wt_m1[l], m1_b + (size_t)l * 3072, nullptr, mlp, nullptr, nullptr,
            M, 3072, D, 0, 0);
        // x = x + g2 * (mlp @ m2 + b)
        gemm_bf16_kernel<3><<<dim3(M / 128, D / 128), 256, 0, stream>>>(
            mlp, wt_m2[l], m2_b + (size_t)l * D, x, nullptr, x, mods,
            M, D, 3072, 4608, 3840);
    }

    // --- final adaLN + projection to 3 ---
    small_mm_kernel<true><<<(Bz * 1536 + 255) / 256, 256, 0, stream>>>(
        adaLN, DA, fl_ada_W, fl_ada_b, fm, Bz, DA, 1536);
    ln_kernel<true, false, true><<<M, 256, 0, stream>>>(
        x, D, nullptr, nullptr, fm, 1536, 0, 768, NA, nullptr, hF);
    final_out_kernel<<<(M * 3 + 255) / 256, 256, 0, stream>>>(
        hF, fl_out_W, fl_out_b, (float*)d_out);
}
